// SwinStyleAttention_69982197121198
// MI455X (gfx1250) — matrix operand order repaired and checked by reading, not compiled, not run on hardware
//
#include <hip/hip_runtime.h>
#include <hip/hip_bf16.h>

typedef _Float16 f16;
typedef __attribute__((ext_vector_type(16))) _Float16 v16h;
typedef __attribute__((ext_vector_type(8)))  _Float16 v8h;
typedef __attribute__((ext_vector_type(8)))  float    v8f;
typedef __attribute__((ext_vector_type(4)))  float    v4f;
typedef __attribute__((ext_vector_type(4)))  unsigned int v4u;
typedef __attribute__((ext_vector_type(8)))  int v8i;
typedef __attribute__((ext_vector_type(4)))  int v4i;

#define SSHIFT 4
#define L      64
#define CCH    128
#define NWIN   8192
#define ATTN_SCALE 0.17677669529663687f   // 32^-0.5

// d_ws layout in f16 elements
#define XWIN_OFF   0                          // [NWIN][64][128] f16
#define XWIN_ELEMS (NWIN * L * CCH)
#define WQKV_OFF   XWIN_ELEMS                 // [384][128] f16 (N-major / transposed)
#define WQKV_ELEMS (384 * 128)
#define WPROJ_OFF  (WQKV_OFF + WQKV_ELEMS)    // [128][128] f16 (N-major / transposed)

// LDS map (f16 elems): q[64][144] @0, k[64][144] @9216, vT[128][72] @18432,
// xbuf[64][72] @27648 (TDM dest, byte offset 55296).
// P[4][64][72] overlays q+k; O[64][144] overlays vT.
#define LDS_HALFS   32256
#define XBUF_OFF_H  27648
#define XBUF_OFF_B  55296

// ---- WMMA fragment helpers (wave32, 16x16x32 f16) ----
__device__ __forceinline__ v16h load_a16(const f16* tile, int ld) {
  int lane = threadIdx.x & 31;
  const f16* p = tile + (lane & 15) * ld + ((lane >> 4) << 3);
  union { v16h v; v8h h[2]; } u;
  u.h[0] = *(const v8h*)(p);
  u.h[1] = *(const v8h*)(p + 16);
  return u.v;
}
__device__ __forceinline__ v16h load_bT16(const f16* bt, int ld) {
  int lane = threadIdx.x & 31;
  const f16* p = bt + (lane & 15) * ld + ((lane >> 4) << 3);
  union { v16h v; v8h h[2]; } u;
  u.h[0] = *(const v8h*)(p);
  u.h[1] = *(const v8h*)(p + 16);
  return u.v;
}
__device__ __forceinline__ void store_c_f16(f16* tile, int ld, v8f c) {
  int lane = threadIdx.x & 31;
  int n  = lane & 15;
  int r0 = (lane >> 4) << 3;
#pragma unroll
  for (int r = 0; r < 8; ++r) tile[(r0 + r) * ld + n] = (f16)c[r];
}
__device__ __forceinline__ void store_cT(f16* dst, int ld, int chan0, int tok0, v8f c) {
  int lane = threadIdx.x & 31;
  int chan = chan0 + (lane & 15);
  int r0   = (lane >> 4) << 3;
  v8h h;
#pragma unroll
  for (int i = 0; i < 8; ++i) h[i] = (f16)c[i];
  *(v8h*)(dst + chan * ld + tok0 + r0) = h;
}

#define WMMA(a, b, c) __builtin_amdgcn_wmma_f32_16x16x32_f16(false, (a), false, (b), (short)0, (c), false, false)

__global__ __launch_bounds__(256) void prep_weights(const float* __restrict__ wqkv,
                                                    const float* __restrict__ wproj,
                                                    f16* __restrict__ wqkvT,
                                                    f16* __restrict__ wprojT) {
  int total = 384 * 128 + 128 * 128;
  for (int i = blockIdx.x * 256 + threadIdx.x; i < total; i += gridDim.x * 256) {
    if (i < 384 * 128) {
      int nn = i >> 7, kk = i & 127;
      wqkvT[i] = (f16)wqkv[kk * 384 + nn];
    } else {
      int j = i - 384 * 128;
      int nn = j >> 7, kk = j & 127;
      wprojT[j] = (f16)wproj[kk * 128 + nn];
    }
  }
}

__global__ __launch_bounds__(256) void prep_xwin(const float* __restrict__ x,
                                                 f16* __restrict__ xwin) {
  __shared__ __align__(16) f16 tile[128 * 72];
  int win = blockIdx.x;
  int b = win >> 10, wy = (win & 1023) >> 5, wx = win & 31;
  for (int i = threadIdx.x; i < 8192; i += 256) {
    int c = i >> 6, t = i & 63;
    int h = ((wy << 3) + (t >> 3) + SSHIFT) & 255;
    int w = ((wx << 3) + (t & 7) + SSHIFT) & 255;
    tile[c * 72 + t] = (f16)x[(((size_t)b * CCH + c) * 256 + h) * 256 + w];
  }
  __syncthreads();
  for (int i = threadIdx.x; i < 8192; i += 256) {
    int t = i >> 7, c = i & 127;
    xwin[(size_t)win * (L * CCH) + i] = tile[c * 72 + t];
  }
}

__global__ __launch_bounds__(256) void swin_attn_kernel(
    const f16* __restrict__ xwin, const f16* __restrict__ wqkvT,
    const f16* __restrict__ wprojT, const float* __restrict__ bproj,
    float* __restrict__ out) {
  __shared__ __align__(16) f16 lds[LDS_HALFS];
  f16* qbuf = lds;                  // [64][144]
  f16* kbuf = lds + 9216;           // [64][144]
  f16* vT   = lds + 18432;          // [128][72]
  f16* xbuf = lds + XBUF_OFF_H;     // [64][72]  (TDM destination)
  f16* pbuf = lds;                  // [4][64][72] overlays q+k
  f16* obuf = lds + 18432;          // [64][144]  overlays vT

  const int win    = blockIdx.x;
  const int wave_s = __builtin_amdgcn_readfirstlane(threadIdx.x) >> 5;  // SGPR wave id
  const int lane   = threadIdx.x & 31;
  const f16* xw = xwin + (size_t)win * (L * CCH);

  // ---------- QKV: [64,128] x [128,384]; K split in two 64-wide halves ----------
  v8f qacc[3][4] = {};
#pragma unroll
  for (int kh = 0; kh < 2; ++kh) {
    __syncthreads();   // previous xbuf contents fully consumed
    if (wave_s == 0) {
      // TDM: DMA X[:, kh*64 .. kh*64+63] (f16, row stride 128) into xbuf with
      // LDS padding: 32 dwords (64 halfs) then +4 dwords -> row stride 72 halfs.
      unsigned long long ga = (unsigned long long)(uintptr_t)(xw + kh * 64);
      v4u g0 = { 1u,                                   // count=1
                 (unsigned)XBUF_OFF_B,                 // lds_addr (bytes)
                 (unsigned)(ga & 0xffffffffu),         // global_addr[31:0]
                 ((unsigned)(ga >> 32) & 0x01ffffffu) | 0x80000000u };  // addr[56:32] | type=2
      v8i g1 = { (1 << 16) | (1 << 20) | (4 << 22) | (3 << 25),  // 2B elems, pad: 32dw interval, 4dw amount
                 (64 << 16),     // tensor_dim0 = 64
                 (64 << 16),     // tensor_dim1 = 64
                 (64 << 16),     // tile_dim0 = 64
                 64,             // tile_dim1 = 64
                 128,            // tensor_dim0_stride = 128
                 0, 0 };
      v4i g2 = { 0, 0, 0, 0 };
      v4i g3 = { 0, 0, 0, 0 };
      v8i g4 = { 0, 0, 0, 0, 0, 0, 0, 0 };
      __builtin_amdgcn_tensor_load_to_lds(g0, g1, g2, g3, g4, 0);
      __builtin_amdgcn_s_wait_tensorcnt(0);
    }
    __syncthreads();   // xbuf ready

    // hoist all B fragments for this wave's 3 N-tiles (this K half)
    v16h bq[3][2];
#pragma unroll
    for (int j = 0; j < 3; ++j) {
      int nq = wave_s * 3 + j;
      const f16* wb = wqkvT + nq * 16 * 128 + kh * 64;
      bq[j][0] = load_bT16(wb + 0, 128);
      bq[j][1] = load_bT16(wb + 32, 128);
    }
#pragma unroll
    for (int mt = 0; mt < 4; ++mt) {
      v16h a0 = load_a16(xbuf + mt * 16 * 72 + 0, 72);
      v16h a1 = load_a16(xbuf + mt * 16 * 72 + 32, 72);
#pragma unroll
      for (int j = 0; j < 3; ++j) {
        qacc[j][mt] = WMMA(a0, bq[j][0], qacc[j][mt]);
        qacc[j][mt] = WMMA(a1, bq[j][1], qacc[j][mt]);
      }
    }
  }
  // scatter q/k/v to LDS (scalar-uniform role branches)
#pragma unroll
  for (int j = 0; j < 3; ++j) {
    int nq = wave_s * 3 + j;
#pragma unroll
    for (int mt = 0; mt < 4; ++mt) {
      if (nq < 8)       store_c_f16(qbuf + mt * 16 * 144 + nq * 16, 144, qacc[j][mt]);
      else if (nq < 16) store_c_f16(kbuf + mt * 16 * 144 + (nq - 8) * 16, 144, qacc[j][mt]);
      else              store_cT(vT, 72, (nq - 16) * 16, mt * 16, qacc[j][mt]);
    }
  }
  __syncthreads();

  // ---------- S = q k^T (2 (head,M-tile) units per wave), softmax ----------
  v8f s[2][4];
#pragma unroll
  for (int u = 0; u < 2; ++u) {
    int unit = wave_s * 2 + u;
    int head = unit >> 2, mt = unit & 3;
    v16h aq = load_a16(qbuf + mt * 16 * 144 + head * 32, 144);
#pragma unroll
    for (int nt = 0; nt < 4; ++nt) {
      v16h bk = load_bT16(kbuf + nt * 16 * 144 + head * 32, 144);
      v8f z = {};
      s[u][nt] = WMMA(aq, bk, z);
    }
#pragma unroll
    for (int r = 0; r < 8; ++r) {
      float m = fmaxf(fmaxf(s[u][0][r], s[u][1][r]), fmaxf(s[u][2][r], s[u][3][r]));
      m = fmaxf(m, __shfl_xor(m, 1));
      m = fmaxf(m, __shfl_xor(m, 2));
      m = fmaxf(m, __shfl_xor(m, 4));
      m = fmaxf(m, __shfl_xor(m, 8));
      float sum = 0.f;
#pragma unroll
      for (int nt = 0; nt < 4; ++nt) {
        float e = __expf((s[u][nt][r] - m) * ATTN_SCALE);
        s[u][nt][r] = e;
        sum += e;
      }
      sum += __shfl_xor(sum, 1);
      sum += __shfl_xor(sum, 2);
      sum += __shfl_xor(sum, 4);
      sum += __shfl_xor(sum, 8);
      float rinv = __builtin_amdgcn_rcpf(sum);
#pragma unroll
      for (int nt = 0; nt < 4; ++nt) s[u][nt][r] *= rinv;
    }
  }
  __syncthreads();   // q,k fully consumed -> P may overlay
#pragma unroll
  for (int u = 0; u < 2; ++u) {
    int unit = wave_s * 2 + u;
    int head = unit >> 2, mt = unit & 3;
#pragma unroll
    for (int nt = 0; nt < 4; ++nt)
      store_c_f16(pbuf + head * 64 * 72 + mt * 16 * 72 + nt * 16, 72, s[u][nt]);
  }
  __syncthreads();

  // ---------- O = P @ v ----------
  v8f oacc[2][2] = {};
#pragma unroll
  for (int u = 0; u < 2; ++u) {
    int unit = wave_s * 2 + u;
    int head = unit >> 2, mt = unit & 3;
#pragma unroll
    for (int ks = 0; ks < 2; ++ks) {
      v16h ap = load_a16(pbuf + head * 64 * 72 + mt * 16 * 72 + ks * 32, 72);
#pragma unroll
      for (int n2 = 0; n2 < 2; ++n2) {
        v16h bv = load_bT16(vT + (head * 32 + n2 * 16) * 72 + ks * 32, 72);
        oacc[u][n2] = WMMA(ap, bv, oacc[u][n2]);
      }
    }
  }
  __syncthreads();   // vT fully consumed -> O may overlay
#pragma unroll
  for (int u = 0; u < 2; ++u) {
    int unit = wave_s * 2 + u;
    int head = unit >> 2, mt = unit & 3;
#pragma unroll
    for (int n2 = 0; n2 < 2; ++n2)
      store_c_f16(obuf + mt * 16 * 144 + head * 32 + n2 * 16, 144, oacc[u][n2]);
  }
  __syncthreads();

  // ---------- proj: [64,128] x [128,128] + bias, N-tile = wave ----------
  {
    int nt = wave_s;
    int n  = lane & 15;
    int c  = nt * 16 + n;
    float bias = bproj[c];
    v16h bw0 = load_bT16(wprojT + nt * 16 * 128 + 0, 128);
    v16h bw1 = load_bT16(wprojT + nt * 16 * 128 + 32, 128);
    v16h bw2 = load_bT16(wprojT + nt * 16 * 128 + 64, 128);
    v16h bw3 = load_bT16(wprojT + nt * 16 * 128 + 96, 128);
    int b  = win >> 10;
    int wy = (win & 1023) >> 5;
    int wx = win & 31;
    int r0 = (lane >> 4) << 3;
#pragma unroll
    for (int mt = 0; mt < 4; ++mt) {
      const f16* ot = obuf + mt * 16 * 144;
      v8f acc = {};
      acc = WMMA(load_a16(ot + 0, 144), bw0, acc);
      acc = WMMA(load_a16(ot + 32, 144), bw1, acc);
      acc = WMMA(load_a16(ot + 64, 144), bw2, acc);
      acc = WMMA(load_a16(ot + 96, 144), bw3, acc);
      // each lane's 8 values: one (b,c,h) row, 8 consecutive w
      int t0 = mt * 16 + r0;                       // multiple of 8 -> col 0
      int h  = ((wy << 3) + (t0 >> 3) + SSHIFT) & 255;
      if (wx != 31) {                              // uniform branch: no w wrap
        float* dst = out + (((size_t)b * CCH + c) * 256 + h) * 256 + ((wx << 3) + SSHIFT);
        v4f lo = { acc[0] + bias, acc[1] + bias, acc[2] + bias, acc[3] + bias };
        v4f hi = { acc[4] + bias, acc[5] + bias, acc[6] + bias, acc[7] + bias };
        *(v4f*)(dst)     = lo;
        *(v4f*)(dst + 4) = hi;
      } else {
#pragma unroll
        for (int r = 0; r < 8; ++r) {
          int w = ((wx << 3) + r + SSHIFT) & 255;
          out[(((size_t)b * CCH + c) * 256 + h) * 256 + w] = acc[r] + bias;
        }
      }
    }
  }
}

extern "C" void kernel_launch(void* const* d_in, const int* in_sizes, int n_in,
                              void* d_out, int out_size, void* d_ws, size_t ws_size,
                              hipStream_t stream) {
  const float* x     = (const float*)d_in[0];
  const float* wqkv  = (const float*)d_in[1];
  const float* wproj = (const float*)d_in[2];
  const float* bproj = (const float*)d_in[3];
  f16* ws     = (f16*)d_ws;
  f16* xwin   = ws + XWIN_OFF;
  f16* wqkvT  = ws + WQKV_OFF;
  f16* wprojT = ws + WPROJ_OFF;

  prep_weights<<<64, 256, 0, stream>>>(wqkv, wproj, wqkvT, wprojT);
  prep_xwin<<<NWIN, 256, 0, stream>>>(x, xwin);
  swin_attn_kernel<<<NWIN, 256, 0, stream>>>(xwin, wqkvT, wprojT, bproj, (float*)d_out);
}
